// ContentConcatAttention_8057358647688
// MI455X (gfx1250) — matrix operand order repaired and checked by reading, not compiled, not run on hardware
//
#include <hip/hip_runtime.h>
#include <hip/hip_bf16.h>

#define B_  32
#define T_  1024
#define DC_ 512
#define DI_ 512
#define DH_ 1024

#define APITCH 40   // bf16 elems per LDS row (80B: 16B-aligned, bank-spread)
#define BPITCH 40

typedef __attribute__((ext_vector_type(16))) __bf16 v16bf;
typedef __attribute__((ext_vector_type(8)))  __bf16 v8bf;
typedef __attribute__((ext_vector_type(8)))  float  v8f;
typedef int v4i __attribute__((vector_size(4 * sizeof(int))));

// ---------------------------------------------------------------------------
// Async global->LDS staging (CDNA5 GLOBAL_LOAD_ASYNC_TO_LDS_B128, ASYNCcnt).
// ---------------------------------------------------------------------------
#if __has_builtin(__builtin_amdgcn_global_load_async_to_lds_b128)
#define ASYNC_OK 1
#else
#define ASYNC_OK 0
#endif

__device__ __forceinline__ void cp_tile_b128(__bf16* l, const __bf16* g) {
#if ASYNC_OK
  __builtin_amdgcn_global_load_async_to_lds_b128(
      (__attribute__((address_space(1))) v4i*)g,
      (__attribute__((address_space(3))) v4i*)l, /*offset=*/0, /*cpol=*/0);
#else
  *reinterpret_cast<v8bf*>(l) = *reinterpret_cast<const v8bf*>(g);
#endif
}

template <int N> __device__ __forceinline__ void wait_async() {
#if ASYNC_OK
#if __has_builtin(__builtin_amdgcn_s_wait_asynccnt)
  __builtin_amdgcn_s_wait_asynccnt((short)N);
#else
  asm volatile("s_wait_asynccnt %0" ::"n"(N) : "memory");
#endif
#endif
}

// gfx1250 has native V_TANH_F32; use it if clang exposes it.
__device__ __forceinline__ float fast_tanh(float z) {
#if __has_builtin(__builtin_amdgcn_tanhf)
  return __builtin_amdgcn_tanhf(z);
#elif __has_builtin(__builtin_amdgcn_tanh_f32)
  return __builtin_amdgcn_tanh_f32(z);
#else
  z = fminf(fmaxf(z, -10.f), 10.f);  // tanh saturates; avoid inf/inf
  const float e = __expf(2.f * z);
  return (e - 1.f) / (e + 1.f);
#endif
}

// ---------------------------------------------------------------------------
// P0a: context fp32 -> bf16 (same layout).  8 elems/thread.
// ---------------------------------------------------------------------------
__global__ __launch_bounds__(256) void cvt_ctx_kernel(const float* __restrict__ ctx,
                                                      __bf16* __restrict__ ctxb) {
  const size_t i = ((size_t)blockIdx.x * 256 + threadIdx.x) * 8;
  const float4* s = reinterpret_cast<const float4*>(ctx + i);
  float4 f0 = s[0], f1 = s[1];
  v8bf o;
  o[0] = (__bf16)f0.x; o[1] = (__bf16)f0.y; o[2] = (__bf16)f0.z; o[3] = (__bf16)f0.w;
  o[4] = (__bf16)f1.x; o[5] = (__bf16)f1.y; o[6] = (__bf16)f1.z; o[7] = (__bf16)f1.w;
  *reinterpret_cast<v8bf*>(ctxb + i) = o;
}

// ---------------------------------------------------------------------------
// P0b: W1 lower half [f][h] -> bf16 TRANSPOSED W1t[h][f] (B-tile order).
// ---------------------------------------------------------------------------
__global__ __launch_bounds__(256) void w1t_kernel(const float* __restrict__ W1,
                                                  __bf16* __restrict__ W1t) {
  __shared__ float tile[32][33];
  const int f0 = blockIdx.x * 32, h0 = blockIdx.y * 32;
  const int tx = threadIdx.x & 31, ty = threadIdx.x >> 5;
  for (int r = ty; r < 32; r += 8)
    tile[r][tx] = W1[(size_t)(f0 + r) * DH_ + h0 + tx];
  __syncthreads();
  for (int r = ty; r < 32; r += 8)
    W1t[(size_t)(h0 + r) * DC_ + f0 + tx] = (__bf16)tile[tx][r];
}

// ---------------------------------------------------------------------------
// K1: xw[b,h] = sum_f x[b,f] * W1[DC+f, h]
// ---------------------------------------------------------------------------
__global__ __launch_bounds__(256) void xw_kernel(const float* __restrict__ x,
                                                 const float* __restrict__ W1,
                                                 float* __restrict__ xw) {
  const int b = blockIdx.x;
  __shared__ float xs[DI_];
  for (int i = threadIdx.x; i < DI_; i += 256) xs[i] = x[b * DI_ + i];
  __syncthreads();
  const int h = threadIdx.x;
  float a0 = 0.f, a1 = 0.f, a2 = 0.f, a3 = 0.f;
  for (int f = 0; f < DI_; ++f) {
    const float xv = xs[f];
    const float* row = W1 + (size_t)(DC_ + f) * DH_;
    a0 += xv * row[h];
    a1 += xv * row[h + 256];
    a2 += xv * row[h + 512];
    a3 += xv * row[h + 768];
  }
  xw[b * DH_ + h]       = a0;
  xw[b * DH_ + h + 256] = a1;
  xw[b * DH_ + h + 512] = a2;
  xw[b * DH_ + h + 768] = a3;
}

// ---------------------------------------------------------------------------
// K2: fused GEMM + tanh + W2-reduce, async double-buffered staging.
// Block tile M=64(t) x N=128(h), K=512 in steps of 32. 8 waves = 4(M) x 2(N).
// ---------------------------------------------------------------------------
__global__ __launch_bounds__(256) void score_gemm_kernel(
    const __bf16* __restrict__ ctxb, const __bf16* __restrict__ W1t,
    const float* __restrict__ W2, const float* __restrict__ xw,
    float* __restrict__ scorePart) {
  const int t0    = blockIdx.x * 64;
  const int hBase = blockIdx.y * 128;
  const int b     = blockIdx.z;
  const int tid   = threadIdx.x;
  const int lane  = tid & 31;
  const int wid   = tid >> 5;
  const int mWave = (wid & 3) * 16;
  const int nWave = (wid >> 2) * 64;

  __shared__ __bf16 As[2][64 * APITCH];
  __shared__ __bf16 Bs[2][128 * BPITCH];
  __shared__ float  scoreS[64];

  if (tid < 64) scoreS[tid] = 0.f;

  v8f acc[4];
#pragma unroll
  for (int j = 0; j < 4; ++j)
#pragma unroll
    for (int r = 0; r < 8; ++r) acc[j][r] = 0.f;

  // staging slices: A 64x32 (1 x b128/thread), B 128x32 transposed (2 x b128)
  const int mA    = tid >> 2;
  const int kcA   = (tid & 3) * 8;
  const int nB    = tid & 127;
  const int kHalf = (tid >> 7) * 16;

  const __bf16* gA = ctxb + ((size_t)b * T_ + t0 + mA) * DC_ + kcA;
  const __bf16* gB = W1t + (size_t)(hBase + nB) * DC_ + kHalf;
  __bf16* lA[2] = {&As[0][mA * APITCH + kcA], &As[1][mA * APITCH + kcA]};
  __bf16* lB[2] = {&Bs[0][nB * BPITCH + kHalf], &Bs[1][nB * BPITCH + kHalf]};

  // one K-step of WMMAs out of buffer `buf`
  auto compute = [&](int buf) {
    const __bf16* AsB = As[buf];
    const __bf16* BsB = Bs[buf];
    union { v16bf v; v8bf h[2]; } aU;
    {  // A 16x32: lane L -> row m, K chunks kb and kb+16
      const __bf16* base = &AsB[(mWave + (lane & 15)) * APITCH];
      const int kb = (lane >> 4) * 8;
      aU.h[0] = *reinterpret_cast<const v8bf*>(base + kb);
      aU.h[1] = *reinterpret_cast<const v8bf*>(base + kb + 16);
    }
#pragma unroll
    for (int j = 0; j < 4; ++j) {  // B 32x16: lane L -> col n, K chunks kb2 and kb2+16
      union { v16bf v; v8bf h[2]; } bU;
      const __bf16* base = &BsB[(nWave + j * 16 + (lane & 15)) * BPITCH];
      const int kb2 = (lane >> 4) * 8;
      bU.h[0] = *reinterpret_cast<const v8bf*>(base + kb2);
      bU.h[1] = *reinterpret_cast<const v8bf*>(base + kb2 + 16);
      acc[j] = __builtin_amdgcn_wmma_f32_16x16x32_bf16(
          false, aU.v, false, bU.v, (short)0, acc[j], false, false);
    }
  };

  // prologue: stage tile 0 into buffer 0
  cp_tile_b128(lA[0], gA);
  cp_tile_b128(lB[0], gB);
  cp_tile_b128(lB[0] + 8, gB + 8);

  // steady state: every iteration prefetches tile i+1 (no branch in body)
#pragma unroll 2
  for (int i = 0; i < 15; ++i) {
    const int nxt = (i + 1) & 1;
    const int kk  = (i + 1) * 32;
    cp_tile_b128(lA[nxt], gA + kk);
    cp_tile_b128(lB[nxt], gB + kk);
    cp_tile_b128(lB[nxt] + 8, gB + kk + 8);
    wait_async<3>();   // in-order completion => tile i landed
    __syncthreads();
    compute(i & 1);
    __syncthreads();   // all waves done with buf before it is refilled
  }
  // epilogue tile 15 (buffer 1)
  wait_async<0>();
  __syncthreads();
  compute(1);

  // ---- epilogue: tanh, scale by W2, reduce over this block's 128 h ----
  // C layout: lane L, vgpr r -> m = mWave + 8*(L>=16) + r, n = L&15
  float sum_r[8];
#pragma unroll
  for (int r = 0; r < 8; ++r) sum_r[r] = 0.f;
#pragma unroll
  for (int j = 0; j < 4; ++j) {
    const int h     = hBase + nWave + j * 16 + (lane & 15);
    const float xwv = xw[b * DH_ + h];
    const float w2v = W2[h];
#pragma unroll
    for (int r = 0; r < 8; ++r)
      sum_r[r] += w2v * fast_tanh(acc[j][r] + xwv);
  }
#pragma unroll
  for (int off = 1; off < 16; off <<= 1)
#pragma unroll
    for (int r = 0; r < 8; ++r)
      sum_r[r] += __shfl_xor(sum_r[r], off);
  if ((lane & 15) == 0) {
    const int mh = mWave + (lane >> 4) * 8;
#pragma unroll
    for (int r = 0; r < 8; ++r)
      atomicAdd(&scoreS[mh + r], sum_r[r]);  // exactly 2 adds/slot: deterministic
  }
  __syncthreads();
  if (tid < 64)
    scorePart[((size_t)blockIdx.y * B_ + b) * T_ + t0 + tid] = scoreS[tid];
}

// ---------------------------------------------------------------------------
// K3: sum the 8 h-chunk partials (fixed order), softmax over T, write attn.
// ---------------------------------------------------------------------------
__global__ __launch_bounds__(256) void softmax_kernel(
    const float* __restrict__ scorePart, float* __restrict__ attn) {
  const int b = blockIdx.x;
  const int tid = threadIdx.x;
  __shared__ float sc[T_];
  for (int i = tid; i < T_; i += 256) {
    float s = 0.f;
    for (int p = 0; p < 8; ++p) s += scorePart[((size_t)p * B_ + b) * T_ + i];
    sc[i] = s;
  }
  __syncthreads();
  float m = -3.4e38f;
  for (int i = tid; i < T_; i += 256) m = fmaxf(m, sc[i]);
  for (int off = 16; off; off >>= 1) m = fmaxf(m, __shfl_xor(m, off));
  __shared__ float wmax[8];
  if ((tid & 31) == 0) wmax[tid >> 5] = m;
  __syncthreads();
  m = wmax[0];
  for (int w = 1; w < 8; ++w) m = fmaxf(m, wmax[w]);

  float s = 0.f;
  for (int i = tid; i < T_; i += 256) {
    const float e = __expf(sc[i] - m);
    sc[i] = e;
    s += e;
  }
  for (int off = 16; off; off >>= 1) s += __shfl_xor(s, off);
  __shared__ float wsum[8];
  if ((tid & 31) == 0) wsum[tid >> 5] = s;
  __syncthreads();
  s = 0.f;
  for (int w = 0; w < 8; ++w) s += wsum[w];
  const float inv = 1.f / s;
  for (int i = tid; i < T_; i += 256) attn[(size_t)b * T_ + i] = sc[i] * inv;
}

// ---------------------------------------------------------------------------
// K4: c[b,d] = sum_t attn[b,t] * context[b,t,d] (fp32 context, no atomics).
// ---------------------------------------------------------------------------
__global__ __launch_bounds__(256) void ctx_reduce_kernel(
    const float* __restrict__ attn, const float* __restrict__ context,
    float* __restrict__ cOut) {
  const int b    = blockIdx.y;
  const int d0   = blockIdx.x * 128;
  const int dl   = threadIdx.x & 127;
  const int half = threadIdx.x >> 7;
  __shared__ float attnS[T_];
  for (int i = threadIdx.x; i < T_; i += 256) attnS[i] = attn[(size_t)b * T_ + i];
  __syncthreads();
  float s = 0.f;
  for (int t = half; t < T_; t += 2)
    s += attnS[t] * context[((size_t)b * T_ + t) * DC_ + d0 + dl];
  __shared__ float red[256];
  red[threadIdx.x] = s;
  __syncthreads();
  if (half == 0) cOut[(size_t)b * DC_ + d0 + dl] = red[dl] + red[dl + 128];
}

// ---------------------------------------------------------------------------
extern "C" void kernel_launch(void* const* d_in, const int* in_sizes, int n_in,
                              void* d_out, int out_size, void* d_ws, size_t ws_size,
                              hipStream_t stream) {
  const float* x       = (const float*)d_in[0];
  // d_in[1] = w : unused by the module
  const float* context = (const float*)d_in[2];
  const float* W1      = (const float*)d_in[3];
  const float* W2      = (const float*)d_in[4];

  float* c_out    = (float*)d_out;              // [B, DC]
  float* attn_out = (float*)d_out + B_ * DC_;   // [B, T]

  char* ws = (char*)d_ws;
  __bf16* ctxb = (__bf16*)ws;                                  // 33.55 MB
  __bf16* W1t  = (__bf16*)(ws + (size_t)B_ * T_ * DC_ * 2);    // 1 MB
  float*  xw   = (float*)(ws + (size_t)B_ * T_ * DC_ * 2 + (size_t)DH_ * DC_ * 2);
  float*  scorePart = xw + B_ * DH_;                           // [8, B, T]

  cvt_ctx_kernel<<<(B_ * T_ * DC_) / (256 * 8), 256, 0, stream>>>(context, ctxb);
  w1t_kernel<<<dim3(DC_ / 32, DH_ / 32), 256, 0, stream>>>(W1, W1t);
  xw_kernel<<<B_, 256, 0, stream>>>(x, W1, xw);
  score_gemm_kernel<<<dim3(T_ / 64, DH_ / 128, B_), 256, 0, stream>>>(
      ctxb, W1t, W2, xw, scorePart);
  softmax_kernel<<<B_, 256, 0, stream>>>(scorePart, attn_out);
  ctx_reduce_kernel<<<dim3(DC_ / 128, B_), 256, 0, stream>>>(attn_out, context, c_out);
}
